// TransformerBlock_63677185130529
// MI455X (gfx1250) — matrix operand order repaired and checked by reading, not compiled, not run on hardware
//
#include <hip/hip_runtime.h>
#include <hip/hip_bf16.h>
#include <math.h>

typedef __bf16 bf16;
typedef __attribute__((ext_vector_type(8)))  bf16  v8bf;
typedef __attribute__((ext_vector_type(16))) bf16  v16bf;
typedef __attribute__((ext_vector_type(8)))  float v8f;
typedef __attribute__((ext_vector_type(4)))  int   v4i;

#define SQRT_2_OVER_PI 0.7978845608028654f

__device__ __forceinline__ float gelu_tanh_f(float x) {
    float t = tanhf(SQRT_2_OVER_PI * (x + 0.044715f * x * x * x));
    return 0.5f * x * (1.0f + t);
}

// ---- 16-byte global->LDS copy: async (CDNA5 GLOBAL_LOAD_ASYNC_TO_LDS_B128) if
// the builtin exists, else synchronous fallback. ----
#if defined(__gfx1250__) && __has_builtin(__builtin_amdgcn_global_load_async_to_lds_b128)
#define HAS_ASYNC_LDS 1
#else
#define HAS_ASYNC_LDS 0
#endif

__device__ __forceinline__ void cp16(const bf16* __restrict__ g, bf16* __restrict__ l) {
#if HAS_ASYNC_LDS
    __builtin_amdgcn_global_load_async_to_lds_b128(
        (__attribute__((address_space(1))) v4i*)(g),
        (__attribute__((address_space(3))) v4i*)(l), 0, 0);
#else
    *(v8bf*)l = *(const v8bf*)g;
#endif
}
// Wait until at most N async copies remain outstanding (per wave, in-order).
template <int N>
__device__ __forceinline__ void cp_wait_n() {
#if HAS_ASYNC_LDS
#if __has_builtin(__builtin_amdgcn_s_wait_asynccnt)
    __builtin_amdgcn_s_wait_asynccnt(N);
#else
    asm volatile("s_wait_asynccnt %0" ::"i"(N) : "memory");
#endif
#endif
}

// ---------------- f32 [R][C] -> bf16 transposed [C][R] (for weights) ----------------
__global__ __launch_bounds__(256) void k_cvtT_bf16(
    const float* __restrict__ src, bf16* __restrict__ dst, int R, int C) {
    __shared__ float t[32][33];
    const int c0 = blockIdx.x * 32, r0 = blockIdx.y * 32;
    const int tx = threadIdx.x & 31, ty = threadIdx.x >> 5;  // ty: 0..7
#pragma unroll
    for (int i = 0; i < 32; i += 8)
        t[ty + i][tx] = src[(size_t)(r0 + ty + i) * C + c0 + tx];
    __syncthreads();
#pragma unroll
    for (int i = 0; i < 32; i += 8)
        dst[(size_t)(c0 + ty + i) * R + r0 + tx] = (bf16)t[tx][ty + i];
}

// ---------------- layernorm (E=1024), one block per row, bf16 out ----------------
__global__ __launch_bounds__(256) void k_layernorm_bf16(
    const float* __restrict__ x, const float* __restrict__ g,
    const float* __restrict__ b, bf16* __restrict__ out) {
    __shared__ float red[256];
    const int tid = threadIdx.x;
    const float* xr = x + (size_t)blockIdx.x * 1024;
    float v[4];
    float s = 0.f;
#pragma unroll
    for (int j = 0; j < 4; ++j) { v[j] = xr[tid + j * 256]; s += v[j]; }
    red[tid] = s; __syncthreads();
    for (int w = 128; w > 0; w >>= 1) { if (tid < w) red[tid] += red[tid + w]; __syncthreads(); }
    const float mu = red[0] * (1.0f / 1024.0f);
    __syncthreads();
    float sq = 0.f;
#pragma unroll
    for (int j = 0; j < 4; ++j) { float d = v[j] - mu; sq += d * d; }
    red[tid] = sq; __syncthreads();
    for (int w = 128; w > 0; w >>= 1) { if (tid < w) red[tid] += red[tid + w]; __syncthreads(); }
    const float rstd = rsqrtf(red[0] * (1.0f / 1024.0f) + 1e-5f);
    bf16* orow = out + (size_t)blockIdx.x * 1024;
#pragma unroll
    for (int j = 0; j < 4; ++j) {
        int c = tid + j * 256;
        orow[c] = (bf16)((v[j] - mu) * rstd * g[c] + b[c]);
    }
}

// ---------------- softmax over rows of length 2048, bf16 out ----------------
__global__ __launch_bounds__(256) void k_softmax_bf16(
    const float* __restrict__ scores, bf16* __restrict__ probs) {
    __shared__ float red[256];
    const int tid = threadIdx.x;
    const float* sr = scores + (size_t)blockIdx.x * 2048;
    bf16* pr = probs + (size_t)blockIdx.x * 2048;
    float v[8];
    float mx = -3.4e38f;
#pragma unroll
    for (int j = 0; j < 8; ++j) { v[j] = sr[tid + j * 256]; mx = fmaxf(mx, v[j]); }
    red[tid] = mx; __syncthreads();
    for (int w = 128; w > 0; w >>= 1) { if (tid < w) red[tid] = fmaxf(red[tid], red[tid + w]); __syncthreads(); }
    mx = red[0];
    __syncthreads();
    float s = 0.f;
#pragma unroll
    for (int j = 0; j < 8; ++j) { v[j] = __expf(v[j] - mx); s += v[j]; }
    red[tid] = s; __syncthreads();
    for (int w = 128; w > 0; w >>= 1) { if (tid < w) red[tid] += red[tid + w]; __syncthreads(); }
    const float inv = 1.0f / red[0];
#pragma unroll
    for (int j = 0; j < 8; ++j) pr[tid + j * 256] = (bf16)(v[j] * inv);
}

// ---------------- tiled bf16 WMMA GEMM (double-buffered async staging) ----------------
// C = epilogue(scale * A@B [+ bias] [+ resid]); outputs f32 and/or bf16.
// A: [M,K] bf16 row-major (lda).  BT==0: B is [K,N] row-major (ldb).
// BT==1: B stored as [N,K] row-major (computes A @ Bstore^T).
// Block: 256 thr = 8 wave32s (2x4); block tile 64x256; wave tile 32x64; BK=32.
enum { EPI_BIAS = 1, EPI_RESID = 2, EPI_GELU = 4, EPI_F32 = 8, EPI_BF16 = 16 };

template <int BT, int FLAGS>
__global__ __launch_bounds__(256) void k_gemm_wmma(
    const bf16* __restrict__ A, int lda, long long strideA,
    const bf16* __restrict__ Bm, int ldb, long long strideB,
    float* __restrict__ Cf, bf16* __restrict__ Cb, int ldc, long long strideC,
    const float* __restrict__ bias,
    const float* __restrict__ resid, int ldr, long long strideR,
    float scale, int K) {
    __shared__ bf16 As[2][64 * 40];    // [64][32+8] padded, double-buffered
    __shared__ bf16 Bt[2][256 * 40];   // transposed B tile [n][k], double-buffered

    const int bz = blockIdx.z;
    A  += (size_t)bz * (size_t)strideA;
    Bm += (size_t)bz * (size_t)strideB;
    if (FLAGS & EPI_F32)   Cf    += (size_t)bz * (size_t)strideC;
    if (FLAGS & EPI_BF16)  Cb    += (size_t)bz * (size_t)strideC;
    if (FLAGS & EPI_RESID) resid += (size_t)bz * (size_t)strideR;

    const int row0 = blockIdx.y * 64;
    const int col0 = blockIdx.x * 256;
    const int tid  = threadIdx.x;
    const int lane = tid & 31;
    const int wave = tid >> 5;
    const int wm = (wave >> 2) * 32;  // 0,32
    const int wn = (wave & 3) * 64;   // 0,64,128,192
    const int lm = lane & 15;         // fragment row (M or N)
    const int kh = lane >> 4;         // lane-half selects K sub-range

    // #async ops issued per thread per tile (in-order completion per wave)
    constexpr int NPEND = (BT == 1) ? 5 : 1;

    auto stage = [&](int kt, int buf) {
        {   // A tile (64x32): one b128 per thread
            int r = tid >> 2;
            int c = (tid & 3) * 8;
            cp16(A + (size_t)(row0 + r) * lda + kt * 32 + c, &As[buf][r * 40 + c]);
        }
        if (BT == 1) {  // B^T stored [N][K]: contiguous rows, 4 async b128 per thread
            int n = tid;
#pragma unroll
            for (int h = 0; h < 4; ++h)
                cp16(Bm + (size_t)(col0 + n) * ldb + kt * 32 + h * 8,
                     &Bt[buf][n * 40 + h * 8]);
        } else {        // B row-major [K][N]: transpose through registers (sync)
            int n0 = (tid & 31) * 8;
            int kb = tid >> 5;  // 0..7
#pragma unroll
            for (int h = 0; h < 4; ++h) {
                int k = kb + h * 8;
                const bf16* gp = Bm + (size_t)(kt * 32 + k) * ldb + col0 + n0;
                v8bf t = *(const v8bf*)gp;
#pragma unroll
                for (int j = 0; j < 8; ++j) Bt[buf][(n0 + j) * 40 + k] = t[j];
            }
        }
    };

    v8f acc[2][4];
#pragma unroll
    for (int i = 0; i < 2; ++i)
#pragma unroll
        for (int j = 0; j < 4; ++j)
#pragma unroll
            for (int r = 0; r < 8; ++r) acc[i][j][r] = 0.0f;

    const int kTiles = K >> 5;
    stage(0, 0);  // prologue

    for (int kt = 0; kt < kTiles; ++kt) {
        const int cur = kt & 1;
        // Prefetch next tile into the other buffer, then wait only for the
        // CURRENT tile's copies (leave the prefetch outstanding).
        if (kt + 1 < kTiles) {
            stage(kt + 1, cur ^ 1);
            cp_wait_n<NPEND>();
        } else {
            cp_wait_n<0>();
        }
        __syncthreads();

        // ---- fragment loads (ISA 16-bit A/B layouts) + 8 WMMAs ----
        union Frag { v16bf v; v8bf h[2]; };
        Frag af[2], bfr[4];
#pragma unroll
        for (int mt = 0; mt < 2; ++mt) {
            // A 16x32: lane half kh holds K = kh*8+0..7 (elems 0-7), 16+kh*8+0..7 (elems 8-15)
            const bf16* p = &As[cur][(wm + mt * 16 + lm) * 40 + kh * 8];
            af[mt].h[0] = *(const v8bf*)p;
            af[mt].h[1] = *(const v8bf*)(p + 16);
        }
#pragma unroll
        for (int nt = 0; nt < 4; ++nt) {
            // B 32x16: lane half kh holds K = kh*8+0..7 (elems 0-7), 16+kh*8+0..7 (elems 8-15), N = lane%16
            const bf16* p = &Bt[cur][(wn + nt * 16 + lm) * 40 + kh * 8];
            bfr[nt].h[0] = *(const v8bf*)p;
            bfr[nt].h[1] = *(const v8bf*)(p + 16);
        }
#pragma unroll
        for (int mt = 0; mt < 2; ++mt)
#pragma unroll
            for (int nt = 0; nt < 4; ++nt)
                acc[mt][nt] = __builtin_amdgcn_wmma_f32_16x16x32_bf16(
                    false, af[mt].v, false, bfr[nt].v, (short)0, acc[mt][nt], false, false);
        __syncthreads();
    }

    // ---- epilogue: C layout = VGPR r -> M = kh*8 + r, N = lane%16 ----
#pragma unroll
    for (int mt = 0; mt < 2; ++mt) {
#pragma unroll
        for (int nt = 0; nt < 4; ++nt) {
            const int n = col0 + wn + nt * 16 + lm;
            const int mbase = row0 + wm + mt * 16 + kh * 8;
            const float bv = (FLAGS & EPI_BIAS) ? bias[n] : 0.0f;
#pragma unroll
            for (int r = 0; r < 8; ++r) {
                const int m = mbase + r;
                float v = acc[mt][nt][r] * scale + bv;
                if (FLAGS & EPI_RESID) v += resid[(size_t)m * ldr + n];
                if (FLAGS & EPI_GELU)  v = gelu_tanh_f(v);
                if (FLAGS & EPI_F32)   Cf[(size_t)m * ldc + n] = v;
                if (FLAGS & EPI_BF16)  Cb[(size_t)m * ldc + n] = (bf16)v;
            }
        }
    }
}

extern "C" void kernel_launch(void* const* d_in, const int* in_sizes, int n_in,
                              void* d_out, int out_size, void* d_ws, size_t ws_size,
                              hipStream_t stream) {
    (void)in_sizes; (void)n_in; (void)out_size; (void)ws_size;
    const int B = 2, S = 2048, E = 1024;
    const int BS = B * S;  // 4096

    const float* x      = (const float*)d_in[0];
    const float* w_qkv  = (const float*)d_in[1];
    const float* b_qkv  = (const float*)d_in[2];
    const float* w_proj = (const float*)d_in[3];
    const float* b_proj = (const float*)d_in[4];
    const float* g1     = (const float*)d_in[5];
    const float* beta1  = (const float*)d_in[6];
    const float* g2     = (const float*)d_in[7];
    const float* beta2  = (const float*)d_in[8];
    const float* w_fc   = (const float*)d_in[9];
    const float* b_fc   = (const float*)d_in[10];
    const float* w_out  = (const float*)d_in[11];
    const float* b_out  = (const float*)d_in[12];
    float* y = (float*)d_out;

    // ---- workspace carve-up ----
    char* ws = (char*)d_ws;
    size_t off = 0;
    auto carve = [&](size_t bytes) -> void* {
        void* p = ws + off;
        off = (off + bytes + 255) & ~(size_t)255;
        return p;
    };
    bf16* wqkvT  = (bf16*)carve((size_t)E * 3 * E * sizeof(bf16));   // [3E][E]
    bf16* wprojT = (bf16*)carve((size_t)E * E * sizeof(bf16));       // [E][E]
    bf16* wfcT   = (bf16*)carve((size_t)E * 4 * E * sizeof(bf16));   // [4E][E]
    bf16* woutT  = (bf16*)carve((size_t)4 * E * E * sizeof(bf16));   // [E][4E]
    bf16* h_bf   = (bf16*)carve((size_t)BS * E * sizeof(bf16));      // LN1 out, reused LN2
    bf16* qkv_bf = (bf16*)carve((size_t)BS * 3 * E * sizeof(bf16));
    float* scores = (float*)carve((size_t)B * S * S * sizeof(float));
    bf16* probs   = (bf16*)carve((size_t)B * S * S * sizeof(bf16));
    bf16* attn_bf = (bf16*)carve((size_t)BS * E * sizeof(bf16));
    float* x1     = (float*)carve((size_t)BS * E * sizeof(float));
    bf16* m1_bf   = (bf16*)carve((size_t)BS * 4 * E * sizeof(bf16));

    const dim3 blk(256);

    // ---- weights -> bf16, transposed so all weight GEMMs take the BT=1 path ----
    k_cvtT_bf16<<<dim3(3 * E / 32, E / 32), blk, 0, stream>>>(w_qkv,  wqkvT,  E, 3 * E);
    k_cvtT_bf16<<<dim3(E / 32,     E / 32), blk, 0, stream>>>(w_proj, wprojT, E, E);
    k_cvtT_bf16<<<dim3(4 * E / 32, E / 32), blk, 0, stream>>>(w_fc,   wfcT,   E, 4 * E);
    k_cvtT_bf16<<<dim3(E / 32, 4 * E / 32), blk, 0, stream>>>(w_out,  woutT,  4 * E, E);

    // ---- h1 = LN1(x) -> bf16 ----
    k_layernorm_bf16<<<dim3(BS), blk, 0, stream>>>(x, g1, beta1, h_bf);

    // ---- qkv = h1 @ w_qkv + b_qkv  -> bf16 [BS, 3E] ----
    k_gemm_wmma<1, EPI_BIAS | EPI_BF16><<<dim3(3 * E / 256, BS / 64, 1), blk, 0, stream>>>(
        h_bf, E, 0, wqkvT, E, 0,
        nullptr, qkv_bf, 3 * E, 0, b_qkv, nullptr, 0, 0, 1.0f, E);

    // ---- scores[b] = Q[b] @ K[b]^T / 8  -> f32 [B, S, S] ----
    k_gemm_wmma<1, EPI_F32><<<dim3(S / 256, S / 64, B), blk, 0, stream>>>(
        qkv_bf + 0, 3 * E, (long long)S * 3 * E,
        qkv_bf + E, 3 * E, (long long)S * 3 * E,
        scores, nullptr, S, (long long)S * S,
        nullptr, nullptr, 0, 0, 0.125f, E);

    // ---- softmax over rows -> bf16 probs ----
    k_softmax_bf16<<<dim3(B * S), blk, 0, stream>>>(scores, probs);

    // ---- attn[b] = probs[b] @ V[b]  -> bf16 [BS, E] ----
    k_gemm_wmma<0, EPI_BF16><<<dim3(E / 256, S / 64, B), blk, 0, stream>>>(
        probs, S, (long long)S * S,
        qkv_bf + 2 * E, 3 * E, (long long)S * 3 * E,
        nullptr, attn_bf, E, (long long)S * E,
        nullptr, nullptr, 0, 0, 1.0f, S);

    // ---- x1 = x + attn @ w_proj + b_proj  -> f32 ----
    k_gemm_wmma<1, EPI_BIAS | EPI_RESID | EPI_F32><<<dim3(E / 256, BS / 64, 1), blk, 0, stream>>>(
        attn_bf, E, 0, wprojT, E, 0,
        x1, nullptr, E, 0, b_proj, x, E, 0, 1.0f, E);

    // ---- h2 = LN2(x1) -> bf16 ----
    k_layernorm_bf16<<<dim3(BS), blk, 0, stream>>>(x1, g2, beta2, h_bf);

    // ---- m1 = gelu(h2 @ w_fc + b_fc) -> bf16 [BS, 4E] ----
    k_gemm_wmma<1, EPI_BIAS | EPI_GELU | EPI_BF16><<<dim3(4 * E / 256, BS / 64, 1), blk, 0, stream>>>(
        h_bf, E, 0, wfcT, E, 0,
        nullptr, m1_bf, 4 * E, 0, b_fc, nullptr, 0, 0, 1.0f, E);

    // ---- y = x1 + m1 @ w_out + b_out -> f32 ----
    k_gemm_wmma<1, EPI_BIAS | EPI_RESID | EPI_F32><<<dim3(E / 256, BS / 64, 1), blk, 0, stream>>>(
        m1_bf, 4 * E, 0, woutT, 4 * E, 0,
        y, nullptr, E, 0, b_out, x1, E, 0, 1.0f, 4 * E);
}
